// AdvancedHybridHOIGNN_67156108640297
// MI455X (gfx1250) — matrix operand order repaired and checked by reading, not compiled, not run on hardware
//
#include <hip/hip_runtime.h>
#include <hip/hip_bf16.h>
#include <cstddef>

// ---------------------------------------------------------------------------
// CDNA5 (gfx1250) wave32 WMMA types
// ---------------------------------------------------------------------------
typedef __attribute__((ext_vector_type(16))) __bf16 v16bf;
typedef __attribute__((ext_vector_type(8)))  __bf16 v8bf;
typedef __attribute__((ext_vector_type(8)))  float  v8f;

__device__ __forceinline__ __bf16 f2bf(float f) { return (__bf16)f; }

__device__ __forceinline__ v8f wmma_bf16(v16bf a, v16bf b, v8f c) {
  // D = A(16x32 bf16) x B(32x16 bf16) + C(16x16 f32)
  return __builtin_amdgcn_wmma_f32_16x16x32_bf16(false, a, false, b,
                                                 (short)0, c, false, false);
}

// A-fragment (16x32): lane -> row M = lane&15; K = {8h..8h+7} U {16+8h..16+8h+7}
// Two contiguous 16-byte bf16 loads per lane.
__device__ __forceinline__ v16bf load_afrag_bf(const __bf16* p, int ld,
                                               int row0, int k0, int lane) {
  int half = lane >> 4, m = lane & 15;
  const __bf16* r = p + (size_t)(row0 + m) * ld + k0 + half * 8;
  v8bf lo = *(const v8bf*)r;
  v8bf hi = *(const v8bf*)(r + 16);
  v16bf a;
#pragma unroll
  for (int i = 0; i < 8; ++i) { a[i] = lo[i]; a[8 + i] = hi[i]; }
  return a;
}

// Row-guarded variant (conv node-axis shifts): clamp row, select zeros.
__device__ __forceinline__ v16bf load_afrag_bf_guard(const __bf16* p, int ld,
                                                     int row0, int k0, int lane,
                                                     int nrows) {
  int half = lane >> 4, m = lane & 15;
  int row = row0 + m;
  bool ok = (row >= 0) && (row < nrows);
  const __bf16* r = p + (size_t)(ok ? row : 0) * ld + k0 + half * 8;
  v8bf lo = *(const v8bf*)r;
  v8bf hi = *(const v8bf*)(r + 16);
  __bf16 z = (__bf16)0.0f;
  v16bf a;
#pragma unroll
  for (int i = 0; i < 8; ++i) {
    a[i]     = ok ? lo[i] : z;
    a[8 + i] = ok ? hi[i] : z;
  }
  return a;
}

// B-fragment (32x16) for C = A @ W^T, W row-major [N,K] bf16: B[k,n]=W[n,k].
// lane -> col n = lane&15; K = {8h..8h+7} U {16+8h..16+8h+7} -> two 16-byte loads.
__device__ __forceinline__ v16bf load_bfrag_bf(const __bf16* W, int ld,
                                               int col0, int k0, int lane) {
  int half = lane >> 4, n = lane & 15;
  const __bf16* r = W + (size_t)(col0 + n) * ld + k0 + half * 8;
  v8bf lo = *(const v8bf*)r;
  v8bf hi = *(const v8bf*)(r + 16);
  v16bf b;
#pragma unroll
  for (int i = 0; i < 8; ++i) { b[i] = lo[i]; b[8 + i] = hi[i]; }
  return b;
}

// ---------------------------------------------------------------------------
// Elementwise / conversion kernels
// ---------------------------------------------------------------------------
__global__ void k_zero(float* p, int n) {
  int i = blockIdx.x * blockDim.x + threadIdx.x;
  if (i < n) p[i] = 0.0f;
}

__global__ void k_cvt(const float* s, __bf16* d, int n) {
  int i = blockIdx.x * blockDim.x + threadIdx.x;
  if (i < n) d[i] = f2bf(s[i]);
}

// conv_w [3 layers][O][I][3taps] -> bf16 [layer][tap][O][I] (contiguous in I)
__global__ void k_convw(const float* w, __bf16* out, int H) {
  int i = blockIdx.x * blockDim.x + threadIdx.x;
  int HH = H * H;
  if (i >= 3 * 3 * HH) return;
  int j  = i / (3 * HH);
  int r  = i % (3 * HH);
  int d  = r / HH;
  int oi = r % HH;
  out[i] = f2bf(w[((size_t)j * HH + oi) * 3 + d]);
}

__global__ void k_deg(const int* dst, float* deg, int E) {
  int e = blockIdx.x * blockDim.x + threadIdx.x;
  if (e < E) atomicAdd(&deg[dst[e]], 1.0f);
}

// scatter-add h[src] into agg[dst]; each thread handles 4 channels
__global__ void k_scatter(const float* h, const int* src, const int* dst,
                          float* agg, int E, int H) {
  int gid = blockIdx.x * blockDim.x + threadIdx.x;
  int chunks = H >> 2;
  int e = gid / chunks;
  if (e >= E) return;
  int c = (gid % chunks) << 2;
  int s = src[e], d = dst[e];
  const float* hp = h + (size_t)s * H + c;
  float* ap = agg + (size_t)d * H + c;
  atomicAdd(ap + 0, hp[0]);
  atomicAdd(ap + 1, hp[1]);
  atomicAdd(ap + 2, hp[2]);
  atomicAdd(ap + 3, hp[3]);
}

// agg(f32, accumulated) / clamp(deg) -> bf16 for the GEMM
__global__ void k_divdeg_bf(const float* agg, const float* deg, __bf16* aggb,
                            int N, int H) {
  int i = blockIdx.x * blockDim.x + threadIdx.x;
  if (i < N * H) aggb[i] = f2bf(agg[i] / fmaxf(deg[i / H], 1.0f));
}

__global__ void k_concat_bf(const __bf16* a, const __bf16* b, __bf16* f,
                            int N, int H) {
  int i = blockIdx.x * blockDim.x + threadIdx.x;
  int H2 = 2 * H;
  if (i >= N * H2) return;
  int row = i / H2, col = i % H2;
  f[i] = (col < H) ? a[(size_t)row * H + col] : b[(size_t)row * H + col - H];
}

// qkv bf16 [N,1536] -> V^T bf16 [head][128][N]  (coalesced writes)
__global__ void k_transpose_v(const __bf16* qkvb, __bf16* vtb, int Nn) {
  int i = blockIdx.x * blockDim.x + threadIdx.x;   // over 512*Nn
  if (i >= 512 * Nn) return;
  int hd_idx = i / Nn;                              // head*128 + d
  int n = i % Nn;
  vtb[i] = qkvb[(size_t)n * 1536 + 1024 + hd_idx];
}

// ---------------------------------------------------------------------------
// Generic WMMA GEMM: C[M,N] = A0 @ W0^T (+ A1 @ W1^T) + bias
// A/W are bf16; writes f32 C and/or bf16 Cb (either may be null).
// One wave per 16x16 tile; blockDim = 256 (8 waves).
// ---------------------------------------------------------------------------
__global__ void k_gemm(const __bf16* A0, const __bf16* W0,
                       const __bf16* A1, const __bf16* W1,
                       const float* bias, float* C, __bf16* Cb,
                       int M, int N, int K) {
  int lane = threadIdx.x & 31;
  int wid = blockIdx.x * (blockDim.x >> 5) + (threadIdx.x >> 5);
  int ntiles = N >> 4;
  int tm = wid / ntiles, tn = wid % ntiles;
  if (tm >= (M >> 4)) return;
  int row0 = tm << 4, col0 = tn << 4;
  v8f acc = {};
  for (int k = 0; k < K; k += 32) {
    v16bf a = load_afrag_bf(A0, K, row0, k, lane);
    v16bf b = load_bfrag_bf(W0, K, col0, k, lane);
    acc = wmma_bf16(a, b, acc);
  }
  if (A1) {
    for (int k = 0; k < K; k += 32) {
      v16bf a = load_afrag_bf(A1, K, row0, k, lane);
      v16bf b = load_bfrag_bf(W1, K, col0, k, lane);
      acc = wmma_bf16(a, b, acc);
    }
  }
  int half = lane >> 4;
  int n = col0 + (lane & 15);
  float bv = bias ? bias[n] : 0.0f;
#pragma unroll
  for (int r = 0; r < 8; ++r) {
    float v = acc[r] + bv;
    size_t idx = (size_t)(row0 + r + 8 * half) * N + n;
    if (C)  C[idx] = v;
    if (Cb) Cb[idx] = f2bf(v);
  }
}

// Conv1d (k=3, pad=1) over node axis as 3 shifted WMMA GEMMs, fused bias+relu.
// W is pre-rearranged bf16 [tap][O][I] (contiguous rows).
__global__ void k_conv(const __bf16* X, const __bf16* W, const float* bias,
                       float* C, int Nn, int H) {
  int lane = threadIdx.x & 31;
  int wid = blockIdx.x * (blockDim.x >> 5) + (threadIdx.x >> 5);
  int ntiles = H >> 4;
  int tm = wid / ntiles, tn = wid % ntiles;
  if (tm >= (Nn >> 4)) return;
  int row0 = tm << 4, col0 = tn << 4;
  v8f acc = {};
  for (int d = 0; d < 3; ++d) {
    const __bf16* Wd = W + (size_t)d * H * H;
    for (int k = 0; k < H; k += 32) {
      v16bf a = load_afrag_bf_guard(X, H, row0 + d - 1, k, lane, Nn);
      v16bf b = load_bfrag_bf(Wd, H, col0, k, lane);
      acc = wmma_bf16(a, b, acc);
    }
  }
  int half = lane >> 4;
  int n = col0 + (lane & 15);
  float bv = bias[n];
#pragma unroll
  for (int r = 0; r < 8; ++r)
    C[(size_t)(row0 + r + 8 * half) * H + n] = fmaxf(acc[r] + bv, 0.0f);
}

// ---------------------------------------------------------------------------
// LayerNorm (in-place on f32, one block per row, D = 256 or 512)
// also writes a bf16 shadow for downstream WMMA consumers.
// ---------------------------------------------------------------------------
__global__ void k_layernorm(float* x, const float* g, const float* b, int D,
                            __bf16* xb) {
  __shared__ float s1[256];
  __shared__ float s2[256];
  int row = blockIdx.x, t = threadIdx.x;
  float* p = x + (size_t)row * D;
  __bf16* pb = xb + (size_t)row * D;
  float x0 = p[t], x1 = 0.0f;
  float a = x0, q = x0 * x0;
  if (D > 256) { x1 = p[t + 256]; a += x1; q += x1 * x1; }
  s1[t] = a; s2[t] = q; __syncthreads();
  for (int s = 128; s > 0; s >>= 1) {
    if (t < s) { s1[t] += s1[t + s]; s2[t] += s2[t + s]; }
    __syncthreads();
  }
  float mean = s1[0] / D;
  float inv = rsqrtf(s2[0] / D - mean * mean + 1e-5f);
  float y0 = (x0 - mean) * inv * g[t] + b[t];
  p[t] = y0; pb[t] = f2bf(y0);
  if (D > 256) {
    float y1 = (x1 - mean) * inv * g[t + 256] + b[t + 256];
    p[t + 256] = y1; pb[t + 256] = f2bf(y1);
  }
}

// SAGE epilogue: h = relu( use_ln ? LN(h) : h ) + res  (in-place, D=256)
// writes f32 h (next layer's residual/scatter source) + bf16 shadow (GEMM A).
__global__ void k_sage_post(float* h, const float* res, const float* g,
                            const float* b, int D, int use_ln, __bf16* hb) {
  __shared__ float s1[256];
  __shared__ float s2[256];
  int row = blockIdx.x, t = threadIdx.x;
  float* p = h + (size_t)row * D;
  float v = p[t];
  if (use_ln) {
    s1[t] = v; s2[t] = v * v; __syncthreads();
    for (int s = 128; s > 0; s >>= 1) {
      if (t < s) { s1[t] += s1[t + s]; s2[t] += s2[t + s]; }
      __syncthreads();
    }
    float mean = s1[0] / D;
    float inv = rsqrtf(s2[0] / D - mean * mean + 1e-5f);
    v = (v - mean) * inv * g[t] + b[t];
  }
  float y = fmaxf(v, 0.0f) + res[(size_t)row * D + t];
  p[t] = y;
  hb[(size_t)row * D + t] = f2bf(y);
}

// ---------------------------------------------------------------------------
// Flash attention: 4 heads, hd=128, one wave per (head, 16-query tile).
// S^T = K_tile @ Q^T puts score tiles directly in A-fragment layout for P @ V.
// Q/K read from qkv bf16; V read from pre-transposed vtb [head][128][N].
// ---------------------------------------------------------------------------
__global__ void k_attn(const __bf16* qkvb, const __bf16* vtb, __bf16* out,
                       int Nn) {
  const int LD = 1536, HD = 128;
  int lane = threadIdx.x & 31;
  int wid = blockIdx.x * (blockDim.x >> 5) + (threadIdx.x >> 5);
  int qtiles = Nn >> 4;
  int head = wid / qtiles;
  int tq = wid % qtiles;
  if (head >= 4) return;
  int q0 = tq << 4;
  int half = lane >> 4, ln = lane & 15;
  const __bf16* Q  = qkvb + head * HD;
  const __bf16* Km = qkvb + 512 + head * HD;
  const __bf16* VT = vtb + (size_t)head * HD * Nn;   // [128][Nn]

  v16bf qf[4];
#pragma unroll
  for (int kk = 0; kk < 4; ++kk)
    qf[kk] = load_bfrag_bf(Q, LD, q0, kk * 32, lane);  // B = Q^T (lane = query)

  const float scale = 0.08838834764831845f;            // 1/sqrt(128)
  float mrun = -1e30f, lrun = 0.0f;
  v8f zero = {};
  v8f oacc[8];
#pragma unroll
  for (int t = 0; t < 8; ++t) oacc[t] = zero;

  for (int j = 0; j < Nn; j += 32) {
    v8f st0 = {}, st1 = {};
#pragma unroll
    for (int kk = 0; kk < 4; ++kk) {
      v16bf ka0 = load_afrag_bf(Km, LD, j,      kk * 32, lane);
      v16bf ka1 = load_afrag_bf(Km, LD, j + 16, kk * 32, lane);
      st0 = wmma_bf16(ka0, qf[kk], st0);
      st1 = wmma_bf16(ka1, qf[kk], st1);
    }
    float p0[8], p1[8];
    float lmax = -1e30f;
#pragma unroll
    for (int r = 0; r < 8; ++r) {
      p0[r] = st0[r] * scale; p1[r] = st1[r] * scale;
      lmax = fmaxf(lmax, fmaxf(p0[r], p1[r]));
    }
    lmax = fmaxf(lmax, __shfl_xor(lmax, 16, 32));      // other 16 keys, same query
    float mnew = fmaxf(mrun, lmax);
    float corr = __expf(mrun - mnew);
    mrun = mnew;
    float lsum = 0.0f;
#pragma unroll
    for (int r = 0; r < 8; ++r) {
      p0[r] = __expf(p0[r] - mnew);
      p1[r] = __expf(p1[r] - mnew);
      lsum += p0[r] + p1[r];
    }
    lsum += __shfl_xor(lsum, 16, 32);
    lrun = lrun * corr + lsum;
    // rescale accumulators: row m = r + 8*half -> corr held by lane m
#pragma unroll
    for (int r = 0; r < 8; ++r) {
      float cf = __shfl(corr, r + 8 * half, 32);
#pragma unroll
      for (int t = 0; t < 8; ++t) oacc[t][r] *= cf;
    }
    // P tile is already in A-fragment layout (keys: {8h+i} U {16+8h+i})
    v16bf pa;
#pragma unroll
    for (int i = 0; i < 8; ++i) { pa[i] = f2bf(p0[i]); pa[8 + i] = f2bf(p1[i]); }
#pragma unroll
    for (int t = 0; t < 8; ++t) {
      v16bf vb = load_bfrag_bf(VT, Nn, t * 16, j, lane);  // contiguous V^T rows
      oacc[t] = wmma_bf16(pa, vb, oacc[t]);
    }
  }
#pragma unroll
  for (int r = 0; r < 8; ++r) {
    float linv = 1.0f / __shfl(lrun, r + 8 * half, 32);
    int m = q0 + r + 8 * half;
#pragma unroll
    for (int t = 0; t < 8; ++t)
      out[(size_t)m * 512 + head * HD + t * 16 + ln] = f2bf(oacc[t][r] * linv);
  }
}

// ---------------------------------------------------------------------------
// Host-side orchestration
// ---------------------------------------------------------------------------
extern "C" void kernel_launch(void* const* d_in, const int* in_sizes, int n_in,
                              void* d_out, int out_size, void* d_ws, size_t ws_size,
                              hipStream_t stream) {
  const int N = 4096, H = 256, L = 6;
  const int E = in_sizes[1] / 2;
  const int HH = H * H;

  const float* x        = (const float*)d_in[0];
  const int*   ei       = (const int*)d_in[1];
  const float* sage_wl  = (const float*)d_in[2];
  const float* sage_wr  = (const float*)d_in[3];
  const float* sage_bl  = (const float*)d_in[4];
  const float* ln_g     = (const float*)d_in[5];
  const float* ln_b     = (const float*)d_in[6];
  const float* conv_w   = (const float*)d_in[7];
  const float* conv_b   = (const float*)d_in[8];
  const float* cnorm_g  = (const float*)d_in[9];
  const float* cnorm_b  = (const float*)d_in[10];
  const float* in_w     = (const float*)d_in[11];
  const float* in_b     = (const float*)d_in[12];
  const float* out_w    = (const float*)d_in[13];
  const float* out_b    = (const float*)d_in[14];
  const float* anorm_g  = (const float*)d_in[15];
  const float* anorm_b  = (const float*)d_in[16];
  const float* fuse_w   = (const float*)d_in[17];
  const float* fuse_b   = (const float*)d_in[18];
  float* outp = (float*)d_out;

  const size_t NH = (size_t)N * H;

  // ---- f32 scratch region ----
  float* ws   = (float*)d_ws;
  float* deg  = ws;                 // N
  float* h0   = deg + N;            // NH
  float* h1   = h0 + NH;            // NH
  float* agg  = h1 + NH;            // NH
  float* c0   = agg + NH;           // NH
  float* c1   = c0 + NH;            // NH
  float* proj = c1 + NH;            // 2*NH (N x 512)
  float* fend = proj + 2 * NH;

  // ---- bf16 scratch region ----
  __bf16* bb     = (__bf16*)fend;
  __bf16* xb     = bb;              // NH
  __bf16* hb0    = xb + NH;         // NH
  __bf16* hb1    = hb0 + NH;        // NH
  __bf16* aggb   = hb1 + NH;        // NH
  __bf16* cb0    = aggb + NH;       // NH
  __bf16* cb1    = cb0 + NH;        // NH
  __bf16* fusedb = cb1 + NH;        // 2*NH
  __bf16* qkvb   = fusedb + 2 * NH; // 6*NH (N x 1536)
  __bf16* vtb    = qkvb + 6 * NH;   // 2*NH (4 heads x 128 x N)
  __bf16* attnb  = vtb + 2 * NH;    // 2*NH
  __bf16* projb  = attnb + 2 * NH;  // 2*NH
  __bf16* wlb    = projb + 2 * NH;  // 6*HH
  __bf16* wrb    = wlb + 6 * (size_t)HH;
  __bf16* cwb    = wrb + 6 * (size_t)HH;     // 9*HH ([layer][tap][O][I])
  __bf16* inwb   = cwb + 9 * (size_t)HH;     // 1536*512
  __bf16* outwb  = inwb + 1536 * 512;        // 512*512
  __bf16* fusewb = outwb + 512 * 512;        // 64*512

  const int TB = 256;
  auto blocks = [](long long n, int tb) { return (int)((n + tb - 1) / tb); };

  // ---- one-time (per call) weight & input conversion to bf16 ----
  k_cvt<<<blocks(6 * HH, TB), TB, 0, stream>>>(sage_wl, wlb, 6 * HH);
  k_cvt<<<blocks(6 * HH, TB), TB, 0, stream>>>(sage_wr, wrb, 6 * HH);
  k_convw<<<blocks(9 * HH, TB), TB, 0, stream>>>(conv_w, cwb, H);
  k_cvt<<<blocks(1536 * 512, TB), TB, 0, stream>>>(in_w, inwb, 1536 * 512);
  k_cvt<<<blocks(512 * 512, TB), TB, 0, stream>>>(out_w, outwb, 512 * 512);
  k_cvt<<<blocks(64 * 512, TB), TB, 0, stream>>>(fuse_w, fusewb, 64 * 512);
  k_cvt<<<blocks((long long)NH, TB), TB, 0, stream>>>(x, xb, (int)NH);

  // ---- degrees ----
  k_zero<<<blocks(N, TB), TB, 0, stream>>>(deg, N);
  k_deg<<<blocks(E, TB), TB, 0, stream>>>(ei + E, deg, E);

  // ---- 6 SAGE layers ----
  const float*  cur  = x;
  const __bf16* curb = xb;
  for (int i = 0; i < L; ++i) {
    float*  nxt  = (i & 1) ? h1 : h0;
    __bf16* nxtb = (i & 1) ? hb1 : hb0;
    k_zero<<<blocks((long long)NH, TB), TB, 0, stream>>>(agg, (int)NH);
    k_scatter<<<blocks((long long)E * (H / 4), TB), TB, 0, stream>>>(
        cur, ei, ei + E, agg, E, H);
    k_divdeg_bf<<<blocks((long long)NH, TB), TB, 0, stream>>>(agg, deg, aggb, N, H);
    int waves = (N / 16) * (H / 16);
    k_gemm<<<waves / 8, TB, 0, stream>>>(aggb, wlb + (size_t)i * HH,
                                         curb, wrb + (size_t)i * HH,
                                         sage_bl + (size_t)i * H,
                                         nxt, nullptr, N, H, H);
    int use_ln = (i < L - 1);
    k_sage_post<<<N, TB, 0, stream>>>(nxt, cur,
                                      use_ln ? ln_g + (size_t)i * H : nullptr,
                                      use_ln ? ln_b + (size_t)i * H : nullptr,
                                      H, use_ln, nxtb);
    cur = nxt; curb = nxtb;
  }
  const __bf16* gnnb = curb;   // hb1 after 6 layers

  // ---- CNN branch: 3 conv layers ----
  const __bf16* cinb = gnnb;
  float*  cbuf[3]  = { c0, c1, c0 };
  __bf16* cbufb[3] = { cb0, cb1, cb0 };
  for (int j = 0; j < 3; ++j) {
    int waves = (N / 16) * (H / 16);
    k_conv<<<waves / 8, TB, 0, stream>>>(cinb, cwb + (size_t)j * 3 * HH,
                                         conv_b + (size_t)j * H, cbuf[j], N, H);
    k_layernorm<<<N, TB, 0, stream>>>(cbuf[j], cnorm_g + (size_t)j * H,
                                      cnorm_b + (size_t)j * H, H, cbufb[j]);
    cinb = cbufb[j];
  }
  const __bf16* cnnb = cinb;   // cb0

  // ---- fuse inputs + attention ----
  k_concat_bf<<<blocks(2 * (long long)NH, TB), TB, 0, stream>>>(gnnb, cnnb,
                                                                fusedb, N, H);
  {   // qkv = fused @ in_proj_w^T + b  (M=4096, N=1536, K=512) -> bf16 only
    int waves = (N / 16) * (1536 / 16);
    k_gemm<<<waves / 8, TB, 0, stream>>>(fusedb, inwb, nullptr, nullptr, in_b,
                                         nullptr, qkvb, N, 1536, 512);
  }
  k_transpose_v<<<blocks(512 * (long long)N, TB), TB, 0, stream>>>(qkvb, vtb, N);
  {   // flash attention: 4 heads x 256 query tiles = 1024 waves
    int waves = 4 * (N / 16);
    k_attn<<<waves / 8, TB, 0, stream>>>(qkvb, vtb, attnb, N);
  }
  {   // out_proj (M=4096, N=512, K=512) -> f32 (LN needs stats)
    int waves = (N / 16) * (512 / 16);
    k_gemm<<<waves / 8, TB, 0, stream>>>(attnb, outwb, nullptr, nullptr, out_b,
                                         proj, nullptr, N, 512, 512);
  }
  k_layernorm<<<N, TB, 0, stream>>>(proj, anorm_g, anorm_b, 512, projb);
  {   // final fuse (M=4096, N=64, K=512) -> f32 d_out
    int waves = (N / 16) * (64 / 16);
    k_gemm<<<waves / 8, TB, 0, stream>>>(projb, fusewb, nullptr, nullptr, fuse_b,
                                         outp, nullptr, N, 64, 512);
  }
}
